// FlowMatchingODE_24627342475791
// MI455X (gfx1250) — compile-verified, matrix operand order checked by reading, not run on hardware
//
#include <hip/hip_runtime.h>
#include <hip/hip_bf16.h>

typedef _Float16 half_t;
typedef __attribute__((ext_vector_type(16))) _Float16 v16h;
typedef __attribute__((ext_vector_type(8)))  _Float16 v8h;
typedef __attribute__((ext_vector_type(8)))  float    v8f;

#define DIM   64
#define HID   512
#define BATCH 8192

// ---------------------------------------------------------------------------
// WMMA helpers (CDNA5 wave32, V_WMMA_F32_16X16X32_F16)
// ---------------------------------------------------------------------------
static __device__ __forceinline__ v8f wmma_f16(v16h a, v16h b, v8f c) {
  return __builtin_amdgcn_wmma_f32_16x16x32_f16(false, a, false, b, (short)0, c,
                                                false, false);
}

// A fragment from row-major A: lane's 16 halves live at p[0..7] and p[16..23].
static __device__ __forceinline__ v16h load_a16(const half_t* __restrict__ p) {
  v8h lo = *(const v8h*)(p);
  v8h hi = *(const v8h*)(p + 16);
  return __builtin_shufflevector(lo, hi, 0, 1, 2, 3, 4, 5, 6, 7,
                                         8, 9, 10, 11, 12, 13, 14, 15);
}

// B fragment from pre-swizzled storage: fully contiguous 16 halves per lane.
static __device__ __forceinline__ v16h load_b16(const half_t* __restrict__ p) {
  v8h lo = *(const v8h*)(p);
  v8h hi = *(const v8h*)(p + 8);
  return __builtin_shufflevector(lo, hi, 0, 1, 2, 3, 4, 5, 6, 7,
                                         8, 9, 10, 11, 12, 13, 14, 15);
}

// Branch-free fast tanh: tanh(x) = 1 - 2/(1 + 2^(x * 2*log2(e)))
static __device__ __forceinline__ float fast_tanh(float x) {
  float e = __builtin_amdgcn_exp2f(x * 2.8853900817779268f);  // v_exp_f32
  float r = __builtin_amdgcn_rcpf(e + 1.0f);                  // v_rcp_f32
  return 1.0f - 2.0f * r;   // saturates correctly: e->inf => 1, e->0 => -1
}

// ---------------------------------------------------------------------------
// Swizzled-B layout.
// Fragment block for (col-tile ct, k-tile kt) stores, per lane (0..31), the 16
// halves of that lane's 32(K)x16(N) WMMA B-fragment contiguously:
//   Bs[ ((ct*ktiles + kt)*32 + lane)*16 + i ]
// where col = ct*16 + (lane&15), kb = (lane&16)?8:0, kl = kb + (i<8 ? i : i+8).
// ---------------------------------------------------------------------------
__global__ void k_swizzleB(const float* __restrict__ B, half_t* __restrict__ Bs,
                           int K, int N) {
  int idx = blockIdx.x * blockDim.x + threadIdx.x;
  if (idx >= K * N) return;
  const int i    = idx & 15;
  const int lane = (idx >> 4) & 31;
  const int blk  = idx >> 9;
  const int ktiles = K >> 5;
  const int kt = blk & (ktiles - 1);   // ktiles is a power of two here
  const int ct = blk / ktiles;
  const int col = ct * 16 + (lane & 15);
  const int kb  = (lane & 16) ? 8 : 0;
  const int kl  = kb + (i < 8 ? i : i + 8);
  Bs[idx] = (half_t)B[(size_t)(kt * 32 + kl) * N + col];
}

// x -> f16 (A operand stays plain row-major)
__global__ void k_f32_to_f16(const float* __restrict__ src,
                             half_t* __restrict__ dst, int n) {
  int i = blockIdx.x * blockDim.x + threadIdx.x;
  if (i < n) dst[i] = (half_t)src[i];
}

// b1eff[j] = b1[j] + t * W1[64, j]   (fold scalar t into the bias)
__global__ void k_b1eff(const float* __restrict__ W1, const float* __restrict__ b1,
                        const float* __restrict__ t, float* __restrict__ b1eff) {
  int j = blockIdx.x * blockDim.x + threadIdx.x;
  if (j < HID) b1eff[j] = b1[j] + t[0] * W1[DIM * HID + j];
}

// M[j,k] = W2[j,k] * sum_i W1[i,j]*W3[k,i], written directly in swizzled-B order
// (j is the K dimension of GEMM4, k is the N dimension).
__global__ void k_buildM(const float* __restrict__ W1, const float* __restrict__ W2,
                         const float* __restrict__ W3, half_t* __restrict__ Ms) {
  int idx = blockIdx.x * blockDim.x + threadIdx.x;
  if (idx >= HID * HID) return;
  const int j = idx >> 9;
  const int k = idx & (HID - 1);
  float c = 0.f;
#pragma unroll 8
  for (int i = 0; i < DIM; ++i) c += W1[i * HID + j] * W3[k * DIM + i];
  const float m = W2[idx] * c;
  // inverse fragment mapping for K-index j, N-index k
  const int kt = j >> 5;
  const int kl = j & 31;
  const int kb = kl & 8;
  const int i8 = (kl & 7) | ((kl >> 4) << 3);       // element index 0..15
  const int lane = (kb ? 16 : 0) + (k & 15);
  const int ct = k >> 4;
  Ms[((size_t)(ct * (HID >> 5) + kt) * 32 + lane) * 16 + i8] = (half_t)m;
}

// ---------------------------------------------------------------------------
// GEMM + tanh epilogue:  H = tanh(A@Bs + bias),  D = 1 - H*H    (f16 outputs)
// Block = 8 waves, 64x128 block tile, 32x32 wave tile (2x2 WMMA accumulators).
// ---------------------------------------------------------------------------
__global__ void k_gemm_tanh(const half_t* __restrict__ A, const half_t* __restrict__ Bs,
                            const float* __restrict__ bias,
                            half_t* __restrict__ Hout, half_t* __restrict__ Dout,
                            int K, int N) {
  const int lane = threadIdx.x & 31;
  const int w    = threadIdx.x >> 5;
  const int wr   = w >> 2, wc = w & 3;
  const int row0 = blockIdx.x * 64 + wr * 32;
  const int col0 = blockIdx.y * 128 + wc * 32;
  const int ktiles = K >> 5;

  const int arow = lane & 15;
  const int kb   = (lane & 16) ? 8 : 0;
  const half_t* pA0 = A + (size_t)(row0 + arow) * K + kb;
  const half_t* pA1 = A + (size_t)(row0 + 16 + arow) * K + kb;
  const half_t* pB0 = Bs + ((size_t)((col0 >> 4)) * ktiles * 32 + lane) * 16;
  const half_t* pB1 = Bs + ((size_t)((col0 >> 4) + 1) * ktiles * 32 + lane) * 16;

  v8f acc[2][2] = {};
  for (int kt = 0; kt < ktiles; ++kt) {
    v16h a0 = load_a16(pA0 + kt * 32);
    v16h a1 = load_a16(pA1 + kt * 32);
    v16h b0 = load_b16(pB0 + (size_t)kt * 512);
    v16h b1 = load_b16(pB1 + (size_t)kt * 512);
    acc[0][0] = wmma_f16(a0, b0, acc[0][0]);
    acc[0][1] = wmma_f16(a0, b1, acc[0][1]);
    acc[1][0] = wmma_f16(a1, b0, acc[1][0]);
    acc[1][1] = wmma_f16(a1, b1, acc[1][1]);
  }
  const int n  = lane & 15;
  const int mb = (lane & 16) ? 8 : 0;
#pragma unroll
  for (int ti = 0; ti < 2; ++ti)
#pragma unroll
    for (int tj = 0; tj < 2; ++tj)
#pragma unroll
      for (int r = 0; r < 8; ++r) {
        const int m = row0 + ti * 16 + mb + r;
        const int c = col0 + tj * 16 + n;
        const float h = fast_tanh(acc[ti][tj][r] + bias[c]);
        Hout[(size_t)m * N + c] = (half_t)h;
        Dout[(size_t)m * N + c] = (half_t)(1.0f - h * h);
      }
}

// Plain GEMM, f16 out (u = d1 @ M)
__global__ void k_gemm_u(const half_t* __restrict__ A, const half_t* __restrict__ Bs,
                         half_t* __restrict__ U, int K, int N) {
  const int lane = threadIdx.x & 31;
  const int w    = threadIdx.x >> 5;
  const int wr   = w >> 2, wc = w & 3;
  const int row0 = blockIdx.x * 64 + wr * 32;
  const int col0 = blockIdx.y * 128 + wc * 32;
  const int ktiles = K >> 5;

  const int arow = lane & 15;
  const int kb   = (lane & 16) ? 8 : 0;
  const half_t* pA0 = A + (size_t)(row0 + arow) * K + kb;
  const half_t* pA1 = A + (size_t)(row0 + 16 + arow) * K + kb;
  const half_t* pB0 = Bs + ((size_t)((col0 >> 4)) * ktiles * 32 + lane) * 16;
  const half_t* pB1 = Bs + ((size_t)((col0 >> 4) + 1) * ktiles * 32 + lane) * 16;

  v8f acc[2][2] = {};
  for (int kt = 0; kt < ktiles; ++kt) {
    v16h a0 = load_a16(pA0 + kt * 32);
    v16h a1 = load_a16(pA1 + kt * 32);
    v16h b0 = load_b16(pB0 + (size_t)kt * 512);
    v16h b1 = load_b16(pB1 + (size_t)kt * 512);
    acc[0][0] = wmma_f16(a0, b0, acc[0][0]);
    acc[0][1] = wmma_f16(a0, b1, acc[0][1]);
    acc[1][0] = wmma_f16(a1, b0, acc[1][0]);
    acc[1][1] = wmma_f16(a1, b1, acc[1][1]);
  }
  const int n  = lane & 15;
  const int mb = (lane & 16) ? 8 : 0;
#pragma unroll
  for (int ti = 0; ti < 2; ++ti)
#pragma unroll
    for (int tj = 0; tj < 2; ++tj)
#pragma unroll
      for (int r = 0; r < 8; ++r)
        U[(size_t)(row0 + ti * 16 + mb + r) * N + col0 + tj * 16 + n] =
            (half_t)acc[ti][tj][r];
}

// GEMM3: v = h2 @ W3 + b3 -> f32 out. N=64, block = 4 waves, 64x64 tile.
__global__ void k_gemm_v(const half_t* __restrict__ A, const half_t* __restrict__ Bs,
                         const float* __restrict__ b3, float* __restrict__ Vout,
                         int K) {
  const int lane = threadIdx.x & 31;
  const int w    = threadIdx.x >> 5;          // 0..3
  const int wr   = w >> 1, wc = w & 1;
  const int row0 = blockIdx.x * 64 + wr * 32;
  const int col0 = wc * 32;
  const int ktiles = K >> 5;

  const int arow = lane & 15;
  const int kb   = (lane & 16) ? 8 : 0;
  const half_t* pA0 = A + (size_t)(row0 + arow) * K + kb;
  const half_t* pA1 = A + (size_t)(row0 + 16 + arow) * K + kb;
  const half_t* pB0 = Bs + ((size_t)((col0 >> 4)) * ktiles * 32 + lane) * 16;
  const half_t* pB1 = Bs + ((size_t)((col0 >> 4) + 1) * ktiles * 32 + lane) * 16;

  v8f acc[2][2] = {};
  for (int kt = 0; kt < ktiles; ++kt) {
    v16h a0 = load_a16(pA0 + kt * 32);
    v16h a1 = load_a16(pA1 + kt * 32);
    v16h b0 = load_b16(pB0 + (size_t)kt * 512);
    v16h b1 = load_b16(pB1 + (size_t)kt * 512);
    acc[0][0] = wmma_f16(a0, b0, acc[0][0]);
    acc[0][1] = wmma_f16(a0, b1, acc[0][1]);
    acc[1][0] = wmma_f16(a1, b0, acc[1][0]);
    acc[1][1] = wmma_f16(a1, b1, acc[1][1]);
  }
  const int n  = lane & 15;
  const int mb = (lane & 16) ? 8 : 0;
#pragma unroll
  for (int ti = 0; ti < 2; ++ti)
#pragma unroll
    for (int tj = 0; tj < 2; ++tj)
#pragma unroll
      for (int r = 0; r < 8; ++r) {
        const int m = row0 + ti * 16 + mb + r;
        const int c = col0 + tj * 16 + n;
        Vout[(size_t)m * DIM + c] = acc[ti][tj][r] + b3[c];
      }
}

// div_out[row] = -sum_k u[row,k]*d2[row,k].  One wave per row.
__global__ void k_div(const half_t* __restrict__ U, const half_t* __restrict__ D2,
                      float* __restrict__ divOut) {
  const int row  = blockIdx.x * 8 + (threadIdx.x >> 5);
  const int lane = threadIdx.x & 31;
  const half_t* ur = U + (size_t)row * HID;
  const half_t* dr = D2 + (size_t)row * HID;
  float s = 0.f;
#pragma unroll
  for (int i = lane; i < HID; i += 32) s += (float)ur[i] * (float)dr[i];
#pragma unroll
  for (int off = 16; off > 0; off >>= 1) s += __shfl_xor(s, off, 32);
  if (lane == 0) divOut[row] = -s;
}

// ---------------------------------------------------------------------------
// Launch
// ---------------------------------------------------------------------------
extern "C" void kernel_launch(void* const* d_in, const int* in_sizes, int n_in,
                              void* d_out, int out_size, void* d_ws, size_t ws_size,
                              hipStream_t stream) {
  const float* t  = (const float*)d_in[0];
  const float* x  = (const float*)d_in[1];
  const float* W1 = (const float*)d_in[2];  // [65 x 512]
  const float* b1 = (const float*)d_in[3];
  const float* W2 = (const float*)d_in[4];  // [512 x 512]
  const float* b2 = (const float*)d_in[5];
  const float* W3 = (const float*)d_in[6];  // [512 x 64]
  const float* b3 = (const float*)d_in[7];
  float* out = (float*)d_out;               // [8192*64 v] ++ [8192 -div]

  char* ws = (char*)d_ws;
  size_t off = 0;
  half_t* W1s   = (half_t*)(ws + off); off += (size_t)DIM * HID * 2;      // 64KB
  half_t* W2s   = (half_t*)(ws + off); off += (size_t)HID * HID * 2;      // 512KB
  half_t* W3s   = (half_t*)(ws + off); off += (size_t)HID * DIM * 2;      // 64KB
  half_t* Ms    = (half_t*)(ws + off); off += (size_t)HID * HID * 2;      // 512KB
  float*  b1eff = (float*)(ws + off);  off += (size_t)HID * 4;            // 2KB
  half_t* xh    = (half_t*)(ws + off); off += (size_t)BATCH * DIM * 2;    // 1MB
  half_t* h1h   = (half_t*)(ws + off); off += (size_t)BATCH * HID * 2;    // 8MB
  half_t* d1h   = (half_t*)(ws + off); off += (size_t)BATCH * HID * 2;
  half_t* h2h   = (half_t*)(ws + off); off += (size_t)BATCH * HID * 2;
  half_t* d2h   = (half_t*)(ws + off); off += (size_t)BATCH * HID * 2;
  half_t* uh    = (half_t*)(ws + off); off += (size_t)BATCH * HID * 2;

  // --- prep: fragment-swizzled f16 weights + folded bias + trace core M ---
  k_swizzleB<<<(DIM * HID + 255) / 256, 256, 0, stream>>>(W1, W1s, DIM, HID);
  k_swizzleB<<<(HID * HID + 255) / 256, 256, 0, stream>>>(W2, W2s, HID, HID);
  k_swizzleB<<<(HID * DIM + 255) / 256, 256, 0, stream>>>(W3, W3s, HID, DIM);
  k_f32_to_f16<<<(BATCH * DIM + 255) / 256, 256, 0, stream>>>(x, xh, BATCH * DIM);
  k_b1eff<<<2, 256, 0, stream>>>(W1, b1, t, b1eff);
  k_buildM<<<(HID * HID + 255) / 256, 256, 0, stream>>>(W1, W2, W3, Ms);

  // --- GEMM1: h1 = tanh(x@W1a + b1eff), d1 = 1-h1^2 ---
  k_gemm_tanh<<<dim3(BATCH / 64, HID / 128), 256, 0, stream>>>(
      xh, W1s, b1eff, h1h, d1h, DIM, HID);
  // --- GEMM2: h2 = tanh(h1@W2 + b2), d2 = 1-h2^2 ---
  k_gemm_tanh<<<dim3(BATCH / 64, HID / 128), 256, 0, stream>>>(
      h1h, W2s, b2, h2h, d2h, HID, HID);
  // --- GEMM3: v = h2@W3 + b3 -> out ---
  k_gemm_v<<<dim3(BATCH / 64, 1), 128, 0, stream>>>(h2h, W3s, b3, out, HID);
  // --- GEMM4: u = d1@M ---
  k_gemm_u<<<dim3(BATCH / 64, HID / 128), 256, 0, stream>>>(d1h, Ms, uh, HID, HID);
  // --- divergence: -sum(u*d2) per row ---
  k_div<<<BATCH / 8, 256, 0, stream>>>(uh, d2h, out + (size_t)BATCH * DIM);
}
